// Net_49512382988633
// MI455X (gfx1250) — matrix operand order repaired and checked by reading, not compiled, not run on hardware
//
#include <hip/hip_runtime.h>
#include <math.h>

typedef __attribute__((ext_vector_type(16))) _Float16 v16h;
typedef __attribute__((ext_vector_type(8)))  _Float16 v8h;
typedef __attribute__((ext_vector_type(8)))  float    v8f;

#define HDIM 128

// ---------------- init: zero accumulator, deg = 1 (self loop) ----------------
__global__ void k_init(float* __restrict__ acc, float* __restrict__ deg, int N) {
    long tid = (long)blockIdx.x * blockDim.x + threadIdx.x;
    long tot = (long)N * HDIM;
    if (tid < tot) acc[tid] = 0.0f;
    if (tid < N)   deg[tid] = 1.0f;
}

// ---------------- degree count over edges ----------------
__global__ void k_deg(const int* __restrict__ ei, float* __restrict__ deg, int E) {
    int e = blockIdx.x * blockDim.x + threadIdx.x;
    if (e < E) atomicAdd(&deg[ei[E + e]], 1.0f);
}

// ---------------- dinv = rsqrt(deg) (deg >= 1 always) ----------------
__global__ void k_dinv(float* __restrict__ d, int N) {
    int i = blockIdx.x * blockDim.x + threadIdx.x;
    if (i < N) d[i] = rsqrtf(d[i]);
}

// ---------------- EW1 = emb @ W1  (VOCAB x H), tiny ----------------
__global__ void k_ew1(const float* __restrict__ emb, const float* __restrict__ W1,
                      float* __restrict__ ew1, int VOCAB, int EMB) {
    int n = threadIdx.x;           // 0..127
    for (int v = 0; v < VOCAB; ++v) {
        float s = 0.f;
        for (int k = 0; k < EMB; ++k) s += emb[v * EMB + k] * W1[k * HDIM + n];
        ew1[v * HDIM + n] = s;
    }
}

// ---------------- pack W2 (f32, KxN row-major) into WMMA B-operand f16 layout -
// block (kt,nt): lane L holds column n = nt*16 + (L&15), K = kt*32 + (L<16?0:16) + j
__global__ void k_packW2(const float* __restrict__ W2, _Float16* __restrict__ w2b) {
    int tid = blockIdx.x * blockDim.x + threadIdx.x;
    if (tid >= HDIM * HDIM) return;
    int j    = tid & 15;
    int lane = (tid >> 4) & 31;
    int blk  = tid >> 9;
    int kt = blk >> 3, nt = blk & 7;
    int k = kt * 32 + ((lane & 16) ? 16 : 0) + j;
    int n = nt * 16 + (lane & 15);
    w2b[tid] = (_Float16)W2[k * HDIM + n];
}

// ---------------- edge scatter: acc[dst] += row[src] * dinv[s]*dinv[d] --------
// one wave per edge; lane handles 4 contiguous features (float4 gather + 4 atomics)
__global__ void k_scatter_ew1(const int* __restrict__ ei, const int* __restrict__ x,
                              const float* __restrict__ ew1, const float* __restrict__ dinv,
                              float* __restrict__ acc, int E) {
    long tid = (long)blockIdx.x * blockDim.x + threadIdx.x;
    int e = (int)(tid >> 5);
    if (e >= E) return;
    int c = (int)(tid & 31);
    int s = ei[e], d = ei[E + e];
    float nrm = dinv[s] * dinv[d];
    const float4 v = *(const float4*)(ew1 + (long)x[s] * HDIM + 4 * c);
    float* o = acc + (long)d * HDIM + 4 * c;
    atomicAdd(o + 0, v.x * nrm);
    atomicAdd(o + 1, v.y * nrm);
    atomicAdd(o + 2, v.z * nrm);
    atomicAdd(o + 3, v.w * nrm);
}

__global__ void k_scatter_h2(const int* __restrict__ ei, const float* __restrict__ h2pre,
                             const float* __restrict__ dinv, float* __restrict__ acc, int E) {
    long tid = (long)blockIdx.x * blockDim.x + threadIdx.x;
    int e = (int)(tid >> 5);
    if (e >= E) return;
    int c = (int)(tid & 31);
    int s = ei[e], d = ei[E + e];
    float nrm = dinv[s] * dinv[d];
    const float4 v = *(const float4*)(h2pre + (long)s * HDIM + 4 * c);
    float* o = acc + (long)d * HDIM + 4 * c;
    atomicAdd(o + 0, v.x * nrm);
    atomicAdd(o + 1, v.y * nrm);
    atomicAdd(o + 2, v.z * nrm);
    atomicAdd(o + 3, v.w * nrm);
}

// ---- h1 = relu(acc + selfloop + b1) as f16; re-zero acc for the next scatter -
__global__ void k_relu1(float* __restrict__ acc, const float* __restrict__ ew1,
                        const int* __restrict__ x, const float* __restrict__ dinv,
                        const float* __restrict__ b1, _Float16* __restrict__ h1h, int N) {
    long tid = (long)blockIdx.x * blockDim.x + threadIdx.x;
    if (tid >= (long)N * HDIM) return;
    int i = (int)(tid >> 7), f = (int)(tid & 127);
    float a = acc[tid];
    acc[tid] = 0.0f;                          // recycle buffer for layer-2 scatter
    float di = dinv[i];
    float v = a + ew1[(long)x[i] * HDIM + f] * di * di + b1[f];
    h1h[tid] = (_Float16)fmaxf(v, 0.0f);
}

// ---------------- WMMA GEMM: h2pre = h1(f16) @ W2(f16), f32 accum -------------
// wave computes a 16x128 row block: 4 K-steps x 8 N-tiles of v_wmma_f32_16x16x32_f16
__global__ __launch_bounds__(256) void k_gemm(const _Float16* __restrict__ h1h,
                                              const _Float16* __restrict__ w2b,
                                              float* __restrict__ h2pre, int N) {
    __shared__ __align__(32) _Float16 w2s[HDIM * HDIM];   // 32 KB, pre-packed B layout
    {   // cooperative stage: 2048 x 16B
        const uint4* s = (const uint4*)w2b;
        uint4* t = (uint4*)w2s;
        for (int i = threadIdx.x; i < (HDIM * HDIM) / 8; i += 256) t[i] = s[i];
    }
    __syncthreads();

    const int lane = threadIdx.x & 31;
    const int m0 = (blockIdx.x * 8 + (threadIdx.x >> 5)) * 16;
    if (m0 >= N) return;                      // wave-uniform: EXEC stays all-ones

    v8f c[8];
#pragma unroll
    for (int nt = 0; nt < 8; ++nt) c[nt] = (v8f)(0.0f);

    const int r  = m0 + (lane & 15);
    const int kb = (lane < 16) ? 0 : 16;      // A layout: K 0-15 vs 16-31
    const v16h* w2v = (const v16h*)w2s;

#pragma unroll
    for (int kt = 0; kt < 4; ++kt) {
        v16h a;
        const _Float16* ap = h1h + (long)r * HDIM + kt * 32 + kb;
        ((v8h*)&a)[0] = *(const v8h*)(ap);
        ((v8h*)&a)[1] = *(const v8h*)(ap + 8);
#pragma unroll
        for (int nt = 0; nt < 8; ++nt) {
            v16h b = w2v[(kt * 8 + nt) * 32 + lane];   // one contiguous 32B LDS read
            c[nt] = __builtin_amdgcn_wmma_f32_16x16x32_f16(
                false, a, false, b, (short)0, c[nt], false, false);
        }
    }

    // C/D layout: VGPR j -> row m0 + j + (lane>=16 ? 8 : 0), col = nt*16 + (lane&15)
    const int rbase = m0 + ((lane < 16) ? 0 : 8);
    const int col = lane & 15;
#pragma unroll
    for (int nt = 0; nt < 8; ++nt)
#pragma unroll
        for (int j = 0; j < 8; ++j)
            h2pre[(long)(rbase + j) * HDIM + nt * 16 + col] = c[nt][j];
}

// ---- final: h2 = relu(acc + selfloop + b2); out = sigmoid(h2 . W3 + b3) ------
// one wave per node, 4 features per lane, shuffle reduction
__global__ void k_final(const float* __restrict__ acc, const float* __restrict__ h2pre,
                        const float* __restrict__ dinv, const float* __restrict__ b2,
                        const float* __restrict__ W3, const float* __restrict__ b3,
                        float* __restrict__ out, int N) {
    int w = blockIdx.x * 8 + (threadIdx.x >> 5);
    if (w >= N) return;
    int lane = threadIdx.x & 31;
    float di = dinv[w], di2 = di * di;
    long base = (long)w * HDIM + 4 * lane;
    float p = 0.f;
#pragma unroll
    for (int j = 0; j < 4; ++j) {
        float hv = acc[base + j] + h2pre[base + j] * di2 + b2[4 * lane + j];
        hv = fmaxf(hv, 0.f);
        p += hv * W3[4 * lane + j];
    }
#pragma unroll
    for (int off = 16; off > 0; off >>= 1) p += __shfl_xor(p, off, 32);
    if (lane == 0) out[w] = 1.0f / (1.0f + expf(-(p + b3[0])));
}

extern "C" void kernel_launch(void* const* d_in, const int* in_sizes, int n_in,
                              void* d_out, int out_size, void* d_ws, size_t ws_size,
                              hipStream_t stream) {
    const int*   x   = (const int*)d_in[0];
    const int*   ei  = (const int*)d_in[1];
    const float* emb = (const float*)d_in[3];
    const float* W1  = (const float*)d_in[4];
    const float* b1  = (const float*)d_in[5];
    const float* W2  = (const float*)d_in[6];
    const float* b2  = (const float*)d_in[7];
    const float* W3  = (const float*)d_in[8];
    const float* b3  = (const float*)d_in[9];
    float* out = (float*)d_out;

    const int N = in_sizes[0];
    const int E = in_sizes[1] / 2;
    const int H = in_sizes[5];          // 128
    const int EMB = in_sizes[4] / H;    // 10
    const int VOCAB = in_sizes[3] / EMB;

    // workspace carve-up (256B aligned)
    char* ws = (char*)d_ws;
    size_t o = 0;
    auto carve = [&](size_t bytes) { char* p = ws + o; o = (o + bytes + 255) & ~(size_t)255; return p; };
    float*    dinv  = (float*)   carve((size_t)N * 4);
    float*    ew1   = (float*)   carve((size_t)VOCAB * H * 4);
    _Float16* w2b   = (_Float16*)carve((size_t)H * H * 2);
    float*    acc   = (float*)   carve((size_t)N * H * 4);
    _Float16* h1h   = (_Float16*)carve((size_t)N * H * 2);
    float*    h2pre = (float*)   carve((size_t)N * H * 4);
    (void)ws_size;

    const long NH = (long)N * H;
    const long EW = (long)E * 32;

    k_init     <<<(int)((NH + 255) / 256), 256, 0, stream>>>(acc, dinv, N);
    k_deg      <<<(E + 255) / 256, 256, 0, stream>>>(ei, dinv, E);
    k_dinv     <<<(N + 255) / 256, 256, 0, stream>>>(dinv, N);
    k_ew1      <<<1, H, 0, stream>>>(emb, W1, ew1, VOCAB, EMB);
    k_packW2   <<<(H * H + 255) / 256, 256, 0, stream>>>(W2, w2b);
    k_scatter_ew1<<<(int)((EW + 255) / 256), 256, 0, stream>>>(ei, x, ew1, dinv, acc, E);
    k_relu1    <<<(int)((NH + 255) / 256), 256, 0, stream>>>(acc, ew1, x, dinv, b1, h1h, N);
    k_gemm     <<<((N + 15) / 16 + 7) / 8, 256, 0, stream>>>(h1h, w2b, h2pre, N);
    k_scatter_h2<<<(int)((EW + 255) / 256), 256, 0, stream>>>(ei, h2pre, dinv, acc, E);
    k_final    <<<(N + 7) / 8, 256, 0, stream>>>(acc, h2pre, dinv, b2, W3, b3, out, N);
}
